// MultiHeadAttention_65798898974762
// MI455X (gfx1250) — matrix operand order repaired and checked by reading, not compiled, not run on hardware
//
#include <hip/hip_runtime.h>

// ---------------------------------------------------------------------------
// MI455X / gfx1250: wave32, WMMA 16x16x32 bf16 (f32 accumulate), TDM loads
// ---------------------------------------------------------------------------

typedef __bf16 bf16_t;
typedef __bf16 bf16x16 __attribute__((ext_vector_type(16)));
typedef __bf16 bf16x8  __attribute__((ext_vector_type(8)));
typedef float  floatx8 __attribute__((ext_vector_type(8)));

typedef unsigned int uint32x4 __attribute__((ext_vector_type(4)));
typedef int          int32x8  __attribute__((ext_vector_type(8)));
typedef int          int32x4  __attribute__((ext_vector_type(4)));

#define D_MODEL 1024
#define N_HEADS 16
#define D_K     64
#define SEQ     2048
#define BATCH   4
#define LN_EPS  1e-5f

#if defined(__gfx1250__) && __has_builtin(__builtin_amdgcn_tensor_load_to_lds)
#define USE_TDM 1
#else
#define USE_TDM 0
#endif

__device__ __forceinline__ unsigned bfbits(float f) {
    unsigned u = __builtin_bit_cast(unsigned, f);
    u += 0x7FFFu + ((u >> 16) & 1u);                 // round-to-nearest-even
    return u >> 16;
}
__device__ __forceinline__ unsigned pack2bf(float lo, float hi) {
    return bfbits(lo) | (bfbits(hi) << 16);
}
__device__ __forceinline__ bf16_t f2bf(float f) {
    unsigned short h = (unsigned short)bfbits(f);
    return __builtin_bit_cast(bf16_t, h);
}

// One 16x16x32 WMMA step. aptr points at As[row][k0 + half*8] (second K chunk
// at +16 elements, per the 16-bit A-matrix layout); bptr at
// Bs[col][k0 + half*8] (second K chunk at +16 elements, same order as A).
__device__ __forceinline__ floatx8 wmma_tile(const bf16_t* aptr,
                                             const bf16_t* bptr,
                                             floatx8 c) {
    bf16x8 a0 = *(const bf16x8*)(aptr);
    bf16x8 a1 = *(const bf16x8*)(aptr + 16);
    bf16x8 b0 = *(const bf16x8*)(bptr);
    bf16x8 b1 = *(const bf16x8*)(bptr + 16);
    bf16x16 a, b;
#pragma unroll
    for (int i = 0; i < 8; ++i) {
        a[i] = a0[i]; a[i + 8] = a1[i];
        b[i] = b0[i]; b[i + 8] = b1[i];
    }
    return __builtin_amdgcn_wmma_f32_16x16x32_bf16(
        /*neg_a=*/false, a, /*neg_b=*/false, b,
        /*c_mod=*/(short)0, c, /*reuse_a=*/false, /*reuse_b=*/false);
}

#if USE_TDM
// Tensor Data Mover: 2D tile load (dword elements) Global -> LDS.
// D# layout per CDNA5 ISA.
// 6-arg builtin form (clang-23 / therock-10.0 headers).
__device__ __forceinline__ void tdm_load_2d(unsigned lds_byte_addr,
                                            const void* gaddr,
                                            unsigned tile_dw,      // tile_dim0 (dwords/row)
                                            unsigned tile_rows,    // tile_dim1
                                            unsigned stride_dw,    // tensor_dim0_stride
                                            unsigned tensor_rows)  // tensor_dim1
{
    unsigned long long ga = (unsigned long long)(size_t)gaddr;
    uint32x4 g0;
    g0[0] = 1u;                                            // count=1, user D#
    g0[1] = lds_byte_addr;                                 // lds_addr
    g0[2] = (unsigned)ga;                                  // global_addr[31:0]
    g0[3] = (unsigned)((ga >> 32) & 0x01FFFFFFu)           // global_addr[56:32]
          | (2u << 30);                                    // type=2 ("image")
    unsigned td0 = stride_dw;                              // tensor_dim0
    unsigned td1 = tensor_rows;                            // tensor_dim1
    int32x8 g1;
    g1[0] = (int)(2u << 16);                               // data_size = 4 bytes
    g1[1] = (int)((td0 & 0xFFFFu) << 16);                  // tensor_dim0[15:0]
    g1[2] = (int)((td0 >> 16) | ((td1 & 0xFFFFu) << 16));  // td0[31:16], td1[15:0]
    g1[3] = (int)((td1 >> 16) | (tile_dw << 16));          // td1[31:16], tile_dim0
    g1[4] = (int)(tile_rows & 0xFFFFu);                    // tile_dim1, tile_dim2=0
    g1[5] = (int)stride_dw;                                // tensor_dim0_stride[31:0]
    g1[6] = 0;                                             // stride0[47:32], stride1[15:0]
    g1[7] = 0;                                             // stride1[47:16]
    int32x4 z4 = {0, 0, 0, 0};
    int32x8 z8 = {0, 0, 0, 0, 0, 0, 0, 0};
    __builtin_amdgcn_tensor_load_to_lds(g0, g1, z4, z4, z8, 0);
}
#endif

// ---------------------------------------------------------------------------
// Kernel 1: Y = X(8192x1024) * W(1024x1024) + bias, output split into heads
// [B,H,S,Dk] as bf16.  64x64 tile per block, 4 waves, 16x64 strip per wave.
// ---------------------------------------------------------------------------
__global__ __launch_bounds__(128)
void proj_kernel(const float* __restrict__ X, const float* __restrict__ W,
                 const float* __restrict__ bias, bf16_t* __restrict__ outH) {
    __shared__ bf16_t As[64][32];
    __shared__ bf16_t Bs[64][32];     // Bs[n][k]
    const int t = threadIdx.x;
    const int lane = t & 31, wave = t >> 5;
    const int half = lane >> 4, row = lane & 15;
    const int m0 = blockIdx.x * 64;
    const int n0 = blockIdx.y * 64;

    floatx8 acc[4] = {};
    for (int k0 = 0; k0 < D_MODEL; k0 += 32) {
        __syncthreads();
#pragma unroll
        for (int j = 0; j < 8; ++j) {             // As: 64 rows x 16 dwords
            int e = j * 128 + t;
            int r = e >> 4, kp = e & 15;
            const float* src = &X[(size_t)(m0 + r) * D_MODEL + k0 + 2 * kp];
            ((unsigned*)As)[e] = pack2bf(src[0], src[1]);
        }
#pragma unroll
        for (int j = 0; j < 8; ++j) {             // Bs[n][2kp..2kp+1] = W[k][n]
            int e = j * 128 + t;
            int n = e & 63, kp = e >> 6;
            float f0 = W[(size_t)(k0 + 2 * kp) * D_MODEL + n0 + n];
            float f1 = W[(size_t)(k0 + 2 * kp + 1) * D_MODEL + n0 + n];
            ((unsigned*)Bs)[n * 16 + kp] = pack2bf(f0, f1);
        }
        __syncthreads();
        const bf16_t* ap = &As[wave * 16 + row][half * 8];
#pragma unroll
        for (int ct = 0; ct < 4; ++ct)
            acc[ct] = wmma_tile(ap, &Bs[ct * 16 + row][half * 8], acc[ct]);
    }
#pragma unroll
    for (int ct = 0; ct < 4; ++ct) {
        int n = n0 + ct * 16 + row;
        float bv = bias[n];
#pragma unroll
        for (int i = 0; i < 8; ++i) {
            int m = m0 + wave * 16 + half * 8 + i;
            int b = m >> 11, s = m & (SEQ - 1);
            int h = n >> 6, dk = n & (D_K - 1);
            size_t idx = (((size_t)(b * N_HEADS + h) * SEQ) + s) * D_K + dk;
            outH[idx] = f2bf(acc[ct][i] + bv);
        }
    }
}

// ---------------------------------------------------------------------------
// Kernel 2: scores[bh] = Qh[bh] * Kh[bh]^T / 8     (2048x2048 per head-batch)
// Kh's [t,dk] row-major layout is directly the B-fragment LDS layout.
// Whole-K (64) tiles loaded once: one TDM descriptor per operand.
// ---------------------------------------------------------------------------
__global__ __launch_bounds__(128)
void scores_kernel(const bf16_t* __restrict__ Qh, const bf16_t* __restrict__ Kh,
                   float* __restrict__ Sc) {
    __shared__ bf16_t As[64][64];
    __shared__ bf16_t Bs[64][64];
    const int t = threadIdx.x;
    const int lane = t & 31, wave = t >> 5;
    const int half = lane >> 4, row = lane & 15;
    const int m0 = blockIdx.x * 64, n0 = blockIdx.y * 64, bh = blockIdx.z;
    const bf16_t* Q = Qh + (size_t)bh * SEQ * D_K;
    const bf16_t* K = Kh + (size_t)bh * SEQ * D_K;

#if USE_TDM
    if (wave == 0) {
        // 64 rows x 32 dwords (64 bf16), row stride 32 dwords, contiguous in LDS
        tdm_load_2d((unsigned)(size_t)&As[0][0], Q + (size_t)m0 * D_K,
                    32, 64, 32, SEQ);
        tdm_load_2d((unsigned)(size_t)&Bs[0][0], K + (size_t)n0 * D_K,
                    32, 64, 32, SEQ);
        __builtin_amdgcn_s_wait_tensorcnt(0);
    }
#else
#pragma unroll
    for (int j = 0; j < 16; ++j) {                 // 64 rows x 32 dwords each
        int e = j * 128 + t;
        int r = e >> 5, c = e & 31;
        ((unsigned*)As)[e] = ((const unsigned*)Q)[(size_t)(m0 + r) * 32 + c];
        ((unsigned*)Bs)[e] = ((const unsigned*)K)[(size_t)(n0 + r) * 32 + c];
    }
#endif
    __syncthreads();

    floatx8 acc[4] = {};
#pragma unroll
    for (int k0 = 0; k0 < D_K; k0 += 32) {
        const bf16_t* ap = &As[wave * 16 + row][k0 + half * 8];
#pragma unroll
        for (int ct = 0; ct < 4; ++ct)
            acc[ct] = wmma_tile(ap, &Bs[ct * 16 + row][k0 + half * 8], acc[ct]);
    }

    float* out = Sc + (size_t)bh * SEQ * SEQ;
    const float scale = 0.125f;                    // 1/sqrt(64)
#pragma unroll
    for (int ct = 0; ct < 4; ++ct) {
        int n = n0 + ct * 16 + row;
#pragma unroll
        for (int i = 0; i < 8; ++i) {
            int m = m0 + wave * 16 + half * 8 + i;
            out[(size_t)m * SEQ + n] = acc[ct][i] * scale;
        }
    }
}

// ---------------------------------------------------------------------------
// Kernel 3: in-place row softmax over 2048 columns (one block per row)
// ---------------------------------------------------------------------------
__global__ __launch_bounds__(256)
void softmax_kernel(float* __restrict__ P) {
    __shared__ float red[256];
    float* p = P + (size_t)blockIdx.x * SEQ;
    const int t = threadIdx.x;
    float x[8];
    float mx = -1e30f;
#pragma unroll
    for (int j = 0; j < 8; ++j) { x[j] = p[t + j * 256]; mx = fmaxf(mx, x[j]); }
    red[t] = mx; __syncthreads();
    for (int s = 128; s > 0; s >>= 1) {
        if (t < s) red[t] = fmaxf(red[t], red[t + s]);
        __syncthreads();
    }
    mx = red[0]; __syncthreads();
    float sum = 0.f;
#pragma unroll
    for (int j = 0; j < 8; ++j) { x[j] = __expf(x[j] - mx); sum += x[j]; }
    red[t] = sum; __syncthreads();
    for (int s = 128; s > 0; s >>= 1) {
        if (t < s) red[t] += red[t + s];
        __syncthreads();
    }
    float inv = 1.0f / red[0];
#pragma unroll
    for (int j = 0; j < 8; ++j) p[t + j * 256] = x[j] * inv;
}

// ---------------------------------------------------------------------------
// Kernel 4: attn_out[bh] = P[bh](2048x2048,f32) * Vh[bh](2048x64,bf16)
// written directly in [B,S,D_MODEL] layout (f32).
// ---------------------------------------------------------------------------
__global__ __launch_bounds__(128)
void pv_kernel(const float* __restrict__ P, const bf16_t* __restrict__ Vh,
               float* __restrict__ AO) {
    __shared__ bf16_t As[64][32];
    __shared__ bf16_t Bs[64][32];                  // Bs[dk][t]
    const int t = threadIdx.x;
    const int lane = t & 31, wave = t >> 5;
    const int half = lane >> 4, row = lane & 15;
    const int m0 = blockIdx.x * 64, bh = blockIdx.z;
    const float* Pb = P + (size_t)bh * SEQ * SEQ;
    const bf16_t* V = Vh + (size_t)bh * SEQ * D_K;

    floatx8 acc[4] = {};
    for (int k0 = 0; k0 < SEQ; k0 += 32) {
        __syncthreads();
#pragma unroll
        for (int j = 0; j < 8; ++j) {             // As from probabilities
            int e = j * 128 + t;
            int r = e >> 4, kp = e & 15;
            const float* src = &Pb[(size_t)(m0 + r) * SEQ + k0 + 2 * kp];
            ((unsigned*)As)[e] = pack2bf(src[0], src[1]);
        }
#pragma unroll
        for (int j = 0; j < 8; ++j) {             // Bs[dk][2tp..2tp+1] = V[t][dk]
            int e = j * 128 + t;
            int dk = e & 63, tp = e >> 6;
            unsigned lo = __builtin_bit_cast(unsigned short,
                              V[(size_t)(k0 + 2 * tp) * D_K + dk]);
            unsigned hi = __builtin_bit_cast(unsigned short,
                              V[(size_t)(k0 + 2 * tp + 1) * D_K + dk]);
            ((unsigned*)Bs)[dk * 16 + tp] = lo | (hi << 16);
        }
        __syncthreads();
        const bf16_t* ap = &As[wave * 16 + row][half * 8];
#pragma unroll
        for (int ct = 0; ct < 4; ++ct)
            acc[ct] = wmma_tile(ap, &Bs[ct * 16 + row][half * 8], acc[ct]);
    }
    const int b = bh >> 4, h = bh & (N_HEADS - 1);
#pragma unroll
    for (int ct = 0; ct < 4; ++ct) {
        int dk = ct * 16 + row;
#pragma unroll
        for (int i = 0; i < 8; ++i) {
            int s = m0 + wave * 16 + half * 8 + i;
            AO[((size_t)b * SEQ + s) * D_MODEL + h * D_K + dk] = acc[ct][i];
        }
    }
}

// ---------------------------------------------------------------------------
// Kernel 5: R = Qin + AO * Wo + bo     (8192x1024)
// ---------------------------------------------------------------------------
__global__ __launch_bounds__(128)
void out_proj_kernel(const float* __restrict__ AO, const float* __restrict__ Wo,
                     const float* __restrict__ bo, const float* __restrict__ Qin,
                     float* __restrict__ R) {
    __shared__ bf16_t As[64][32];
    __shared__ bf16_t Bs[64][32];
    const int t = threadIdx.x;
    const int lane = t & 31, wave = t >> 5;
    const int half = lane >> 4, row = lane & 15;
    const int m0 = blockIdx.x * 64;
    const int n0 = blockIdx.y * 64;

    floatx8 acc[4] = {};
    for (int k0 = 0; k0 < D_MODEL; k0 += 32) {
        __syncthreads();
#pragma unroll
        for (int j = 0; j < 8; ++j) {
            int e = j * 128 + t;
            int r = e >> 4, kp = e & 15;
            const float* src = &AO[(size_t)(m0 + r) * D_MODEL + k0 + 2 * kp];
            ((unsigned*)As)[e] = pack2bf(src[0], src[1]);
        }
#pragma unroll
        for (int j = 0; j < 8; ++j) {
            int e = j * 128 + t;
            int n = e & 63, kp = e >> 6;
            float f0 = Wo[(size_t)(k0 + 2 * kp) * D_MODEL + n0 + n];
            float f1 = Wo[(size_t)(k0 + 2 * kp + 1) * D_MODEL + n0 + n];
            ((unsigned*)Bs)[n * 16 + kp] = pack2bf(f0, f1);
        }
        __syncthreads();
        const bf16_t* ap = &As[wave * 16 + row][half * 8];
#pragma unroll
        for (int ct = 0; ct < 4; ++ct)
            acc[ct] = wmma_tile(ap, &Bs[ct * 16 + row][half * 8], acc[ct]);
    }
#pragma unroll
    for (int ct = 0; ct < 4; ++ct) {
        int n = n0 + ct * 16 + row;
        float bv = bo[n];
#pragma unroll
        for (int i = 0; i < 8; ++i) {
            int m = m0 + wave * 16 + half * 8 + i;
            size_t idx = (size_t)m * D_MODEL + n;
            R[idx] = acc[ct][i] + bv + Qin[idx];
        }
    }
}

// ---------------------------------------------------------------------------
// Kernel 6: LayerNorm over D_MODEL (one block of 256 per row)
// ---------------------------------------------------------------------------
__global__ __launch_bounds__(256)
void ln_kernel(const float* __restrict__ R, const float* __restrict__ gamma,
               const float* __restrict__ beta, float* __restrict__ Out) {
    __shared__ float red[256];
    const float* r = R + (size_t)blockIdx.x * D_MODEL;
    const int t = threadIdx.x;
    float x[4];
    float sum = 0.f;
#pragma unroll
    for (int j = 0; j < 4; ++j) { x[j] = r[t + j * 256]; sum += x[j]; }
    red[t] = sum; __syncthreads();
    for (int s = 128; s > 0; s >>= 1) {
        if (t < s) red[t] += red[t + s];
        __syncthreads();
    }
    float mu = red[0] * (1.0f / D_MODEL);
    __syncthreads();
    float vs = 0.f;
#pragma unroll
    for (int j = 0; j < 4; ++j) { float d = x[j] - mu; vs += d * d; }
    red[t] = vs; __syncthreads();
    for (int s = 128; s > 0; s >>= 1) {
        if (t < s) red[t] += red[t + s];
        __syncthreads();
    }
    float inv = rsqrtf(red[0] * (1.0f / D_MODEL) + LN_EPS);
    float* o = Out + (size_t)blockIdx.x * D_MODEL;
#pragma unroll
    for (int j = 0; j < 4; ++j) {
        int n = t + j * 256;
        o[n] = (x[j] - mu) * inv * gamma[n] + beta[n];
    }
}

// ---------------------------------------------------------------------------
extern "C" void kernel_launch(void* const* d_in, const int* in_sizes, int n_in,
                              void* d_out, int out_size, void* d_ws, size_t ws_size,
                              hipStream_t stream) {
    const float* q     = (const float*)d_in[0];
    const float* k     = (const float*)d_in[1];
    const float* v     = (const float*)d_in[2];
    const float* Wq    = (const float*)d_in[3];
    const float* bq    = (const float*)d_in[4];
    const float* Wk    = (const float*)d_in[5];
    const float* bk    = (const float*)d_in[6];
    const float* Wv    = (const float*)d_in[7];
    const float* bv    = (const float*)d_in[8];
    const float* Wo    = (const float*)d_in[9];
    const float* bo    = (const float*)d_in[10];
    const float* gamma = (const float*)d_in[11];
    const float* beta  = (const float*)d_in[12];

    float* out  = (float*)d_out;                           // [B,S,D]
    float* attn = out + (size_t)BATCH * SEQ * D_MODEL;     // [B,H,S,S]

    char* ws = (char*)d_ws;
    bf16_t* qh = (bf16_t*)(ws);                            // 16 MB
    bf16_t* kh = (bf16_t*)(ws + (size_t)16 * 1024 * 1024); // 16 MB
    bf16_t* vh = (bf16_t*)(ws + (size_t)32 * 1024 * 1024); // 16 MB
    float*  ao = (float*) (ws + (size_t)48 * 1024 * 1024); // 32 MB
    float*  rr = (float*) (ws + (size_t)80 * 1024 * 1024); // 32 MB

    dim3 gemm_grid(128, 16, 1);      // M=8192/64, N=1024/64
    dim3 gemm_block(128);

    proj_kernel<<<gemm_grid, gemm_block, 0, stream>>>(q, Wq, bq, qh);
    proj_kernel<<<gemm_grid, gemm_block, 0, stream>>>(k, Wk, bk, kh);
    proj_kernel<<<gemm_grid, gemm_block, 0, stream>>>(v, Wv, bv, vh);

    scores_kernel<<<dim3(SEQ / 64, SEQ / 64, BATCH * N_HEADS),
                    gemm_block, 0, stream>>>(qh, kh, attn);

    softmax_kernel<<<dim3(BATCH * N_HEADS * SEQ), dim3(256), 0, stream>>>(attn);

    pv_kernel<<<dim3(SEQ / 64, 1, BATCH * N_HEADS),
                gemm_block, 0, stream>>>(attn, vh, ao);

    out_proj_kernel<<<gemm_grid, gemm_block, 0, stream>>>(ao, Wo, bo, q, rr);

    ln_kernel<<<dim3(BATCH * SEQ), dim3(256), 0, stream>>>(rr, gamma, beta, out);
}
